// Quantization_64269890617731
// MI455X (gfx1250) — matrix operand order repaired and checked by reading, not compiled, not run on hardware
//
#include <hip/hip_runtime.h>
#include <hip/hip_bf16.h>

// Residual VQ, fused over all 4 levels. bf16 WMMA (16x16x32, f32 accum) for
// both the distance GEMM and the softmax@codes GEMM. Codebook (pre-converted
// to bf16 in d_ws) is staged per level into LDS with a single Tensor Data
// Mover descriptor (tensor_load_to_lds) using TDM padding to produce the
// bank-conflict-padded LDS stride. The per-wave x tile is also TDM-loaded.
// Identities: qhard = x - r_final, qsoft = x - t_final,
// (qsoft-qhard)^2 = (r-t)^2, hard_dist level term = sum r^2 after update,
// which doubles as the ||r||^2 needed by the next level's distance GEMM.

#define N_ROWS 262144
#define DIM 128
#define LVL 4
#define KC 256

#define NW 2                               // waves per block
#define ROWS_PER_BLOCK (NW * 16)           // 32 rows
#define NBLK (N_ROWS / ROWS_PER_BLOCK)     // 8192 blocks

#define DP (DIM + 4)                       // padded strides (bank-conflict relief)
#define KP (KC + 4)
#define KE (KC + 8)

// d_ws layout: [0, NBLK*4 floats) block partials; bf16 codebook at WS_CB_OFF
#define WS_CB_OFF (NBLK * 4 * 4)           // bytes (= 128 KB)

typedef __attribute__((ext_vector_type(16))) __bf16 v16bf;
typedef __attribute__((ext_vector_type(8)))  float  v8f;
typedef __attribute__((ext_vector_type(16))) unsigned short u16x16;
typedef __attribute__((ext_vector_type(4)))  unsigned int v4u;
typedef __attribute__((ext_vector_type(4)))  int v4i;
typedef __attribute__((ext_vector_type(8)))  int v8i;

static __device__ __forceinline__ unsigned short f2bf(float f) {
    unsigned u = __builtin_bit_cast(unsigned, f);
    u += 0x7FFFu + ((u >> 16) & 1u);            // round to nearest even
    return (unsigned short)(u >> 16);
}
static __device__ __forceinline__ float bf2f(unsigned short h) {
    return __builtin_bit_cast(float, (unsigned)h << 16);
}

// ---- Tensor Data Mover: 2D tile Global -> LDS with optional LDS padding ----
// Descriptor bitfields per CDNA5 ISA.
// pad_interval_code: pad after 2^(c+1) DWORDs; pad_amount_code: (amount-1) DWORDs.
static __device__ __forceinline__ void tdm_load_2d(
    unsigned lds_addr, unsigned long long gaddr,
    unsigned data_size_code,                 // 0=1B,1=2B,2=4B,3=8B
    unsigned dim0, unsigned dim1,            // tensor dims (elements)
    unsigned tile0, unsigned tile1,          // tile dims (elements)
    unsigned stride0,                        // dim0 stride (elements)
    unsigned pad_interval_code, unsigned pad_amount_code, unsigned pad_en)
{
    v4u g0;
    g0[0] = 1u;                                              // count=1, user mode
    g0[1] = lds_addr;                                        // LDS byte address
    g0[2] = (unsigned)(gaddr & 0xFFFFFFFFull);               // global_addr[31:0]
    g0[3] = (unsigned)((gaddr >> 32) & 0x1FFFFFFu)           // global_addr[56:32]
          | (2u << 30);                                      // type=2 ("image")
    v8i g1;
    g1[0] = (int)((data_size_code << 16) | (pad_en << 20) |
                  (pad_interval_code << 22) | (pad_amount_code << 25));
    g1[1] = (int)(dim0 << 16);                               // tensor_dim0[15:0]
    g1[2] = (int)((dim0 >> 16) | (dim1 << 16));              // dim0 hi | dim1 lo
    g1[3] = (int)((dim1 >> 16) | (tile0 << 16));             // dim1 hi | tile_dim0
    g1[4] = (int)(tile1);                                    // tile_dim1 (tile_dim2=0)
    g1[5] = (int)stride0;                                    // tensor_dim0_stride lo
    g1[6] = 0;                                               // stride hi | dim1_stride lo
    g1[7] = 0;
    v4i z4 = {0, 0, 0, 0};
#if __clang_major__ >= 23
    v8i z8 = {0, 0, 0, 0, 0, 0, 0, 0};
    __builtin_amdgcn_tensor_load_to_lds(g0, g1, z4, z4, z8, 0);
#else
    __builtin_amdgcn_tensor_load_to_lds(g0, g1, z4, z4, 0);
#endif
}

// ---- one-shot f32 -> bf16 codebook conversion into d_ws ----
__global__ __launch_bounds__(256, 1)
void rvq_cvt_cb(const float* __restrict__ cb, unsigned short* __restrict__ cbb)
{
    int i = (blockIdx.x * 256 + threadIdx.x) * 4;            // LVL*KC*DIM total
    float4 v = ((const float4*)cb)[i >> 2];
    cbb[i + 0] = f2bf(v.x); cbb[i + 1] = f2bf(v.y);
    cbb[i + 2] = f2bf(v.z); cbb[i + 3] = f2bf(v.w);
}

__global__ __launch_bounds__(NW * 32, 1)
void rvq_main(const float* __restrict__ x, const unsigned short* __restrict__ cbb,
              float* __restrict__ out, float* __restrict__ ws)
{
    __shared__ unsigned short sCode[KC][DP];     // bf16 codebook level
    __shared__ float          sC2[KC];           // ||c_k||^2 (bf16-consistent)
    __shared__ float          sDist[NW][16][KP]; // d2 tile per wave
    __shared__ unsigned short sE[NW][16][KE];    // exp values (bf16)
    __shared__ float          sR[NW][16][DP];    // residual
    __shared__ float          sT[NW][16][DP];    // x - qsoft_cum
    __shared__ float          sR2[NW][16];       // per-row ||r||^2
    __shared__ float          sInvS[NW][16];     // 1/softmax-sum
    __shared__ int            sArg[NW][16];      // argmin code
    __shared__ float          sSD[NW][16];       // accumulated soft_dist terms
    __shared__ float          sHD[NW][16];       // accumulated hard_dist terms
    __shared__ float          sJC[NW];

    const int tid  = threadIdx.x;
    const int wv   = tid >> 5;
    const int lane = tid & 31;
    const int n    = lane & 15;   // col-in-tile / A-frag row
    const int h    = lane >> 4;   // half-wave
    const int rb   = blockIdx.x * ROWS_PER_BLOCK + wv * 16;

    if (lane < 16) { sSD[wv][n] = 0.f; sHD[wv][n] = 0.f; }

    // ---- TDM: x rows (16x128 f32) -> sR[wv] with padded stride 132 f32 ----
    // row = 128 DWORDs -> pad_interval code 6 (=128 DW), pad 4 DW -> code 3.
    {
        unsigned ldsR = (unsigned)(uintptr_t)&sR[wv][0][0];
        unsigned long long gx = (unsigned long long)(uintptr_t)(x + (size_t)rb * DIM);
        tdm_load_2d(ldsR, gx, /*4B*/2, DIM, 16, DIM, 16, DIM, 6, 3, 1);
        __builtin_amdgcn_s_wait_tensorcnt(0);
    }
    // mirror into t = x, compute initial ||r||^2 = ||x||^2
    for (int i = 0; i < 16; ++i) {
        int q = lane + 32 * i;
        int r = q >> 5, c = (q & 31) * 4;
        sT[wv][r][c + 0] = sR[wv][r][c + 0];
        sT[wv][r][c + 1] = sR[wv][r][c + 1];
        sT[wv][r][c + 2] = sR[wv][r][c + 2];
        sT[wv][r][c + 3] = sR[wv][r][c + 3];
    }
    {
        float s = 0.f;
        #pragma unroll 8
        for (int j = 0; j < 64; ++j) { float a = sR[wv][n][h * 64 + j]; s += a * a; }
        s += __shfl_xor(s, 16, 32);
        if (h == 0) sR2[wv][n] = s;
    }
    __syncthreads();

    int* codeOut = (int*)(out + (size_t)2 * N_ROWS * DIM + 4);

    for (int l = 0; l < LVL; ++l) {
        // ---- TDM: stage bf16 codebook level (256x128) -> sCode, stride 132 ----
        // row = 128 bf16 = 64 DWORDs -> interval code 5; pad 2 DW -> code 1.
        __syncthreads();                 // previous-level readers done with sCode
        if (wv == 0) {
            unsigned ldsC = (unsigned)(uintptr_t)&sCode[0][0];
            unsigned long long gc =
                (unsigned long long)(uintptr_t)(cbb + (size_t)l * KC * DIM);
            tdm_load_2d(ldsC, gc, /*2B*/1, DIM, KC, DIM, KC, DIM, 5, 1, 1);
            __builtin_amdgcn_s_wait_tensorcnt(0);
        }
        __syncthreads();
        for (int k = tid; k < KC; k += NW * 32) {
            float s = 0.f;
            #pragma unroll 8
            for (int d = 0; d < DIM; ++d) { float c = bf2f(sCode[k][d]); s += c * c; }
            sC2[k] = s;
        }
        __syncthreads();

        // ---- GEMM1: dist = r . c^T  (A = residual, B = codes^T) ----
        v16bf a1[4];
        #pragma unroll
        for (int kk = 0; kk < 4; ++kk) {
            u16x16 t;
            #pragma unroll
            for (int e = 0; e < 16; ++e) {
                int k = kk * 32 + h * 16 + e;  // contiguous k = 16h + e (matches B)
                t[e] = f2bf(sR[wv][n][k]);
            }
            a1[kk] = __builtin_bit_cast(v16bf, t);
        }
        for (int ct = 0; ct < 16; ++ct) {
            v8f acc = {0.f, 0.f, 0.f, 0.f, 0.f, 0.f, 0.f, 0.f};
            #pragma unroll
            for (int kk = 0; kk < 4; ++kk) {
                u16x16 tb;
                #pragma unroll
                for (int e = 0; e < 16; ++e)                         // contiguous k = 16h + e (matches A)
                    tb[e] = sCode[ct * 16 + n][kk * 32 + h * 16 + e];
                v16bf b = __builtin_bit_cast(v16bf, tb);
                acc = __builtin_amdgcn_wmma_f32_16x16x32_bf16(
                        false, a1[kk], false, b, (short)0, acc, false, false);
            }
            #pragma unroll
            for (int v = 0; v < 8; ++v) {                            // C layout: M=v+8h
                int row = v + 8 * h;
                int col = ct * 16 + n;
                float d2 = sR2[wv][row] + sC2[col] - 2.f * acc[v];
                sDist[wv][row][col] = fmaxf(d2, 0.f);
            }
        }
        __syncthreads();

        // ---- per-row max d2, argmin d2 ----
        float mx = 0.f, mn = 3.4e38f; int arg = 0;
        {
            int base = h * 128;
            for (int j = 0; j < 128; ++j) {
                float d2 = sDist[wv][n][base + j];
                mx = fmaxf(mx, d2);
                if (d2 < mn) { mn = d2; arg = base + j; }
            }
            float omx = __shfl_xor(mx, 16, 32);
            float omn = __shfl_xor(mn, 16, 32);
            int   oar = __shfl_xor(arg, 16, 32);
            mx = fmaxf(mx, omx);
            if (omn < mn || (omn == mn && oar < arg)) { mn = omn; arg = oar; }
            if (h == 0) {
                sArg[wv][n] = arg;
                codeOut[(size_t)(rb + n) * LVL + l] = arg;
            }
        }
        // ---- softmax numerators e = exp((mn - d2)/mx), sum S ----
        {
            float invmx = mx > 0.f ? 1.f / mx : 0.f;
            float S = 0.f;
            int base = h * 128;
            for (int j = 0; j < 128; ++j) {
                float d2 = sDist[wv][n][base + j];
                float e  = __expf((mn - d2) * invmx);
                S += e;
                sE[wv][n][base + j] = f2bf(e);
            }
            S += __shfl_xor(S, 16, 32);
            if (h == 0) sInvS[wv][n] = 1.f / S;
        }
        __syncthreads();

        // ---- GEMM2: soft = (E @ codes) * invS; update r, t; row stats ----
        v16bf a2[8];
        #pragma unroll
        for (int kc = 0; kc < 8; ++kc) {
            u16x16 t;
            #pragma unroll
            for (int e = 0; e < 16; ++e) {
                int k = kc * 32 + h * 16 + e;
                t[e] = sE[wv][n][k];
            }
            a2[kc] = __builtin_bit_cast(v16bf, t);
        }
        float invs[8]; int bk[8];
        #pragma unroll
        for (int v = 0; v < 8; ++v) {
            int row = v + 8 * h;
            invs[v] = sInvS[wv][row];
            bk[v]   = sArg[wv][row];
        }
        float pt2[8], pr2[8];
        #pragma unroll
        for (int v = 0; v < 8; ++v) { pt2[v] = 0.f; pr2[v] = 0.f; }

        for (int dt = 0; dt < 8; ++dt) {
            v8f acc = {0.f, 0.f, 0.f, 0.f, 0.f, 0.f, 0.f, 0.f};
            #pragma unroll
            for (int kc = 0; kc < 8; ++kc) {
                u16x16 tb;
                #pragma unroll
                for (int e = 0; e < 16; ++e)
                    tb[e] = sCode[kc * 32 + h * 16 + e][dt * 16 + n];
                v16bf b = __builtin_bit_cast(v16bf, tb);
                acc = __builtin_amdgcn_wmma_f32_16x16x32_bf16(
                        false, a2[kc], false, b, (short)0, acc, false, false);
            }
            int col = dt * 16 + n;
            #pragma unroll
            for (int v = 0; v < 8; ++v) {
                int row = v + 8 * h;
                float soft = acc[v] * invs[v];
                float hard = bf2f(sCode[bk[v]][col]);
                float rn = sR[wv][row][col] - hard;
                float tn = sT[wv][row][col] - soft;
                sR[wv][row][col] = rn;
                sT[wv][row][col] = tn;
                pr2[v] += rn * rn;
                pt2[v] += tn * tn;
            }
        }
        #pragma unroll
        for (int v = 0; v < 8; ++v) {
            float a = pr2[v], b = pt2[v];
            #pragma unroll
            for (int m = 1; m < 16; m <<= 1) {
                a += __shfl_xor(a, m, 32);
                b += __shfl_xor(b, m, 32);
            }
            if (n == 0) {
                int row = v + 8 * h;
                sR2[wv][row]  = a;       // ||r||^2 for next level; error at l=3
                sHD[wv][row] += a;       // hard_dist term
                sSD[wv][row] += b;       // soft_dist term
            }
        }
        __syncthreads();
    }

    // ---- outputs: qhard = x - r, qsoft = x - t; joint_center partial ----
    const float4* x4 = (const float4*)x;
    float jc = 0.f;
    for (int i = tid; i < ROWS_PER_BLOCK * DIM / 4; i += NW * 32) {
        int r32 = i >> 5, c4 = i & 31;
        int w2 = r32 >> 4, r16 = r32 & 15;
        size_t grow = (size_t)blockIdx.x * ROWS_PER_BLOCK + r32;
        float4 xv = x4[grow * 32 + c4];
        int c = c4 * 4;
        float r0 = sR[w2][r16][c + 0], r1 = sR[w2][r16][c + 1];
        float r2 = sR[w2][r16][c + 2], r3 = sR[w2][r16][c + 3];
        float t0 = sT[w2][r16][c + 0], t1 = sT[w2][r16][c + 1];
        float t2 = sT[w2][r16][c + 2], t3 = sT[w2][r16][c + 3];
        float4 qh = { xv.x - r0, xv.y - r1, xv.z - r2, xv.w - r3 };
        float4 qs = { xv.x - t0, xv.y - t1, xv.z - t2, xv.w - t3 };
        ((float4*)out)[grow * 32 + c4] = qh;
        ((float4*)(out + (size_t)N_ROWS * DIM))[grow * 32 + c4] = qs;
        float d0 = r0 - t0, d1 = r1 - t1, d2 = r2 - t2, d3 = r3 - t3;
        jc += d0 * d0 + d1 * d1 + d2 * d2 + d3 * d3;
    }
    #pragma unroll
    for (int m = 1; m < 32; m <<= 1) jc += __shfl_xor(jc, m, 32);
    if (lane == 0) sJC[wv] = jc;
    __syncthreads();

    if (tid == 0) {
        float SD = 0.f, HD = 0.f, ER = 0.f, JC = 0.f;
        for (int w = 0; w < NW; ++w) {
            JC += sJC[w];
            for (int r = 0; r < 16; ++r) {
                SD += sSD[w][r]; HD += sHD[w][r]; ER += sR2[w][r];
            }
        }
        float* p = ws + (size_t)blockIdx.x * 4;
        p[0] = SD; p[1] = HD; p[2] = ER; p[3] = JC;
    }
}

__global__ __launch_bounds__(256, 1)
void rvq_finalize(const float* __restrict__ ws, float* __restrict__ out)
{
    __shared__ float red[256][4];
    int t = threadIdx.x;
    float a0 = 0.f, a1 = 0.f, a2 = 0.f, a3 = 0.f;
    for (int b = t; b < NBLK; b += 256) {
        const float* p = ws + (size_t)b * 4;
        a0 += p[0]; a1 += p[1]; a2 += p[2]; a3 += p[3];
    }
    red[t][0] = a0; red[t][1] = a1; red[t][2] = a2; red[t][3] = a3;
    __syncthreads();
    for (int s = 128; s > 0; s >>= 1) {
        if (t < s) {
            red[t][0] += red[t + s][0]; red[t][1] += red[t + s][1];
            red[t][2] += red[t + s][2]; red[t][3] += red[t + s][3];
        }
        __syncthreads();
    }
    if (t == 0) {
        float invN = 1.f / (float)N_ROWS;
        float* sc = out + (size_t)2 * N_ROWS * DIM;
        sc[0] = red[0][0] * invN;                          // soft_dist
        sc[1] = red[0][1] * invN;                          // hard_dist
        sc[2] = red[0][3] / ((float)N_ROWS * (float)DIM);  // joint_center
        sc[3] = red[0][2] * invN;                          // error
    }
}

extern "C" void kernel_launch(void* const* d_in, const int* in_sizes, int n_in,
                              void* d_out, int out_size, void* d_ws, size_t ws_size,
                              hipStream_t stream) {
    (void)in_sizes; (void)n_in; (void)out_size; (void)ws_size;
    const float* x  = (const float*)d_in[0];
    const float* cb = (const float*)d_in[1];
    float* out = (float*)d_out;
    float* ws  = (float*)d_ws;
    unsigned short* cbb = (unsigned short*)((char*)d_ws + WS_CB_OFF);

    // 1) codebook f32 -> bf16 (LVL*KC*DIM = 131072 elems, 4 per thread)
    rvq_cvt_cb<<<(LVL * KC * DIM) / (256 * 4), 256, 0, stream>>>(cb, cbb);
    // 2) fused residual-VQ over all levels (TDM staging + WMMA GEMMs)
    rvq_main<<<NBLK, NW * 32, 0, stream>>>(x, cbb, out, ws);
    // 3) deterministic scalar reduction
    rvq_finalize<<<1, 256, 0, stream>>>(ws, out);
}
